// BiLSTMWithDotAttention_30623116820909
// MI455X (gfx1250) — matrix operand order repaired and checked by reading, not compiled, not run on hardware
//
#include <hip/hip_runtime.h>
#include <math.h>

#define SEQ   1024
#define BATCH 8
#define EMB   300
#define HID   512
#define GATES 2048        // 4*H
#define KTE   10          // k-tiles for embed GEMM (320/32)
#define KTH   16          // k-tiles for hidden GEMM (512/32)
#define MT    512         // m-tiles (8192/16)
#define NT    128         // n-tiles (2048/16)

typedef __attribute__((ext_vector_type(16))) __bf16    v16bf;
typedef __attribute__((ext_vector_type(8)))  float     v8f;
typedef __attribute__((ext_vector_type(8)))  unsigned  v8u;

__device__ __forceinline__ unsigned short f2bf(float f) {
    unsigned u = __builtin_bit_cast(unsigned, f);
    u += 0x7FFFu + ((u >> 16) & 1u);           // RNE
    return (unsigned short)(u >> 16);
}
__device__ __forceinline__ float bf2f(unsigned short h) {
    unsigned u = ((unsigned)h) << 16;
    return __builtin_bit_cast(float, u);
}
__device__ __forceinline__ float sigm(float x) { return 1.0f / (1.0f + __expf(-x)); }

__device__ __forceinline__ v8f vzero() {
    v8f z;
#pragma unroll
    for (int i = 0; i < 8; ++i) z[i] = 0.0f;
    return z;
}

// Fragment stored as 256 dwords per 16x32(bf16) tile: dword index = lane*8 + v.
// Each lane loads 32 contiguous bytes.
__device__ __forceinline__ v16bf ld_frag(const unsigned* base, int lane) {
    const v8u* p = (const v8u*)base;
    return __builtin_bit_cast(v16bf, p[lane]);
}

__device__ __forceinline__ v8f wmma_bf16(v16bf a, v16bf b, v8f c) {
    return __builtin_amdgcn_wmma_f32_16x16x32_bf16(false, a, false, b, (short)0, c, false, false);
}

// A-frag map (16-bit A 16x32): element (row M, kk) -> lane/vgpr/half
//   lane = M + ((kk>>3)&1)*16 ; v = ((kk>>4)&1)*4 + ((kk&7)>>1) ; half = kk&1
// B-frag map (16-bit B 32x16): element (kk, col N) -> lane/vgpr/half
//   lane = N + ((kk>>3)&1)*16 ; v = ((kk>>4)&1)*4 + ((kk&7)>>1) ; half = kk&1

// ---------------- Kernel 1: embedding gather -> A-fragment-packed bf16 x ----------------
__global__ void pack_x_kernel(const int* __restrict__ ids, const float* __restrict__ emb,
                              unsigned* __restrict__ xfrag) {
    int idx = blockIdx.x * 256 + threadIdx.x;   // (mt, kt, dword)
    int mt  = idx / (KTE * 256);
    int rem = idx % (KTE * 256);
    int kt  = rem >> 8;
    int dw  = rem & 255;
    int lane = dw >> 3, v = dw & 7;
    int r = lane & 15, lh = lane >> 4;
    int m = mt * 16 + r;
    int tok = ids[m];
    int kbase = ((v >= 4) ? 16 : 0) + lh * 8 + (v & 3) * 2;  // inverse A-map
    int e0 = kt * 32 + kbase;
    float f0 = (e0     < EMB) ? emb[(size_t)tok * EMB + e0]     : 0.0f;
    float f1 = (e0 + 1 < EMB) ? emb[(size_t)tok * EMB + e0 + 1] : 0.0f;
    xfrag[idx] = (unsigned)f2bf(f0) | ((unsigned)f2bf(f1) << 16);
}

// ---------------- Kernel 2: weight -> B-fragment-packed bf16 ----------------
__global__ void pack_w_kernel(const float* __restrict__ W, int K, int Kt,
                              unsigned* __restrict__ dst) {
    int bx = blockIdx.x;                 // nt*Kt + kt
    int nt = bx / Kt, kt = bx % Kt;
    int dw = threadIdx.x;
    int lane = dw >> 3, v = dw & 7;
    int n  = nt * 16 + (lane & 15);
    int kk = ((v >= 4) ? 16 : 0) + (lane >> 4) * 8 + (v & 3) * 2;   // inverse B-map
    int k  = kt * 32 + kk;
    float f0 = (k     < K) ? W[(size_t)n * K + k]     : 0.0f;
    float f1 = (k + 1 < K) ? W[(size_t)n * K + k + 1] : 0.0f;
    dst[(size_t)bx * 256 + dw] = (unsigned)f2bf(f0) | ((unsigned)f2bf(f1) << 16);
}

// ---------------- Kernel 3: pre = x @ w_ih^T + b_ih + b_hh (both dirs) ----------------
__global__ void gemm_pre_kernel(const unsigned* __restrict__ xfrag,
                                const unsigned* __restrict__ wihfrag,  // [2][NT][KTE][256]
                                const float* __restrict__ bihf, const float* __restrict__ bhhf,
                                const float* __restrict__ bihb, const float* __restrict__ bhhb,
                                unsigned short* __restrict__ prebf) {  // [2][8192][2048] bf16
    int dir  = blockIdx.y;
    int w    = threadIdx.x >> 5, lane = threadIdx.x & 31;
    int l15  = lane & 15, lh = lane >> 4;
    int gw   = blockIdx.x * 8 + w;
    int mt   = gw >> 7, nt = gw & 127;
    int n    = nt * 16 + l15;
    float bsum = dir ? (bihb[n] + bhhb[n]) : (bihf[n] + bhhf[n]);
    v8f acc;
#pragma unroll
    for (int r = 0; r < 8; ++r) acc[r] = bsum;
    const unsigned* wb = wihfrag + ((size_t)dir * NT + nt) * KTE * 256;
#pragma unroll
    for (int kt = 0; kt < KTE; ++kt) {
        v16bf a = ld_frag(xfrag + ((size_t)mt * KTE + kt) * 256, lane);
        v16bf b = ld_frag(wb + kt * 256, lane);
        acc = wmma_bf16(a, b, acc);
    }
    size_t base = (size_t)dir * 8192 * GATES;
#pragma unroll
    for (int r = 0; r < 8; ++r) {
        int m = mt * 16 + r + lh * 8;     // C layout: lanes 0-15 -> M=r, 16-31 -> M=8+r
        prebf[base + (size_t)m * GATES + n] = f2bf(acc[r]);
    }
}

// ---------------- Kernel 4: sequential LSTM scan (1 workgroup per direction) ----------------
__global__ void __launch_bounds__(1024, 1)
lstm_scan_kernel(const unsigned short* __restrict__ prebf,
                 const unsigned* __restrict__ whhfrag,   // [2][NT][KTH][256]
                 unsigned short* __restrict__ outA,      // [8][64][32][512]  A-frag  (M=t, K=d)
                 unsigned short* __restrict__ outBdm,    // [8][64][32][512]  B-frag  (K=d, N=t)
                 unsigned short* __restrict__ outBtm) {  // [8][64][32][512]  B-frag  (K=t, N=d)
    __shared__ unsigned       hfrag[KTH * 256];          // 16 KB: h (16x512) in A-frag layout
    __shared__ unsigned short gbuf[BATCH * GATES];       // 32 KB: gate preactivations (bf16)
    int tid = threadIdx.x;
    int dir = blockIdx.x;
    int w = tid >> 5, lane = tid & 31, l15 = lane & 15, lh = lane >> 4;
    for (int i = tid; i < KTH * 256; i += 1024) hfrag[i] = 0u;
    float cst[4] = {0.f, 0.f, 0.f, 0.f};                 // cell state: (b,j)->thread map is fixed
    __syncthreads();

    for (int t = 0; t < SEQ; ++t) {
        int tr = dir ? (SEQ - 1 - t) : t;
        // ---- g = pre[tr] + h @ whh^T   (this wave owns columns [w*64, w*64+64))
        v8f acc[4];
#pragma unroll
        for (int tl = 0; tl < 4; ++tl) {
            int n = (w * 4 + tl) * 16 + l15;
#pragma unroll
            for (int r = 0; r < 8; ++r) {
                float pv = 0.0f;
                if (lh == 0)   // valid batch rows 0..7 live in lanes 0-15
                    pv = bf2f(prebf[((size_t)dir * 8192 + (size_t)r * SEQ + tr) * GATES + n]);
                acc[tl][r] = pv;
            }
        }
#pragma unroll 4
        for (int kt = 0; kt < KTH; ++kt) {
            v16bf a = ld_frag(hfrag + kt * 256, lane);
#pragma unroll
            for (int tl = 0; tl < 4; ++tl) {
                int nt = w * 4 + tl;
                v16bf b = ld_frag(whhfrag + (((size_t)dir * NT + nt) * KTH + kt) * 256, lane);
                acc[tl] = wmma_bf16(a, b, acc[tl]);
            }
        }
#pragma unroll
        for (int tl = 0; tl < 4; ++tl) {
            int n = (w * 4 + tl) * 16 + l15;
#pragma unroll
            for (int r = 0; r < 8; ++r)
                if (lh == 0) gbuf[r * GATES + n] = f2bf(acc[tl][r]);
        }
        __syncthreads();

        // ---- gates + state update: 4 (b,j) elements per thread
        unsigned short* hf16 = (unsigned short*)hfrag;
#pragma unroll
        for (int q = 0; q < 4; ++q) {
            int e = tid * 4 + q;
            int b = e >> 9, j = e & 511;
            float gi = bf2f(gbuf[b * GATES + j]);
            float gf = bf2f(gbuf[b * GATES + 512 + j]);
            float gg = bf2f(gbuf[b * GATES + 1024 + j]);
            float go = bf2f(gbuf[b * GATES + 1536 + j]);
            float c = sigm(gf) * cst[q] + sigm(gi) * tanhf(gg);
            float h = sigm(go) * tanhf(c);
            cst[q] = c;
            unsigned short hb = f2bf(h);
            // h -> LDS A-frag (M=b, K=j) for next step
            {
                int kt = j >> 5, kk = j & 31;
                int la = b + ((kk >> 3) & 1) * 16;
                int va = ((kk >> 4) & 1) * 4 + ((kk & 7) >> 1);
                hf16[kt * 512 + la * 16 + va * 2 + (kk & 1)] = hb;
            }
            // out[b, tr, d] scattered to three fragment-ordered global copies
            int d   = j + dir * HID;
            int tt  = tr >> 4, rr = tr & 15;
            int ktd = d >> 5, kkd = d & 31;
            {   // A-frag: M = t-in-tile, K = d
                int la = rr + ((kkd >> 3) & 1) * 16;
                int va = ((kkd >> 4) & 1) * 4 + ((kkd & 7) >> 1);
                outA[(((size_t)b * 64 + tt) * 32 + ktd) * 512 + la * 16 + va * 2 + (kkd & 1)] = hb;
            }
            {   // B-frag: K = d, N = t
                int lb = rr + ((kkd >> 3) & 1) * 16;
                int vb = ((kkd >> 4) & 1) * 4 + ((kkd & 7) >> 1);
                outBdm[(((size_t)b * 64 + tt) * 32 + ktd) * 512 + lb * 16 + vb * 2 + (kkd & 1)] = hb;
            }
            {   // B-frag: K = t, N = d
                int ntd = d >> 4, nd = d & 15;
                int ktt = tr >> 5, kkt = tr & 31;
                int lb = nd + ((kkt >> 3) & 1) * 16;
                int vb = ((kkt >> 4) & 1) * 4 + ((kkt & 7) >> 1);
                outBtm[(((size_t)b * 64 + ntd) * 32 + ktt) * 512 + lb * 16 + vb * 2 + (kkt & 1)] = hb;
            }
        }
        __syncthreads();
    }
}

// ---------------- Kernel 5: scores -> softmax -> attention, one 16-row strip per block ----------------
__global__ void __launch_bounds__(256, 1)
attn_kernel(const unsigned* __restrict__ outA, const unsigned* __restrict__ outBdm,
            const unsigned* __restrict__ outBtm, float* __restrict__ out) {
    __shared__ unsigned short sco[16 * 1024];   // 32 KB: scores / exp (bf16)
    __shared__ unsigned wfragd[32 * 256];       // 32 KB: weights in A-frag layout (K=t)
    int tid = threadIdx.x;
    int strip = blockIdx.x;                     // 0..63
    int b = blockIdx.y;                         // 0..7
    int w = tid >> 5, lane = tid & 31, l15 = lane & 15, lh = lane >> 4;

    // ---- phase 1: scores[16 x 1024] = out[strip rows] @ out^T
    v8f acc[8];
#pragma unroll
    for (int j = 0; j < 8; ++j) acc[j] = vzero();
    for (int kd = 0; kd < 32; ++kd) {
        v16bf a = ld_frag(outA + (((size_t)b * 64 + strip) * 32 + kd) * 256, lane);
#pragma unroll
        for (int j = 0; j < 8; ++j) {
            int nt = w * 8 + j;
            v16bf bb = ld_frag(outBdm + (((size_t)b * 64 + nt) * 32 + kd) * 256, lane);
            acc[j] = wmma_bf16(a, bb, acc[j]);
        }
    }
#pragma unroll
    for (int j = 0; j < 8; ++j) {
        int col = (w * 8 + j) * 16 + l15;
#pragma unroll
        for (int r = 0; r < 8; ++r) sco[(r + lh * 8) * 1024 + col] = f2bf(acc[j][r]);
    }
    __syncthreads();

    // ---- phase 2: row softmax; weights -> LDS A-frag (M=row, K=t)
    if (tid < 16) {
        int row = tid;
        float mx = -3.0e38f;
        for (int tc = 0; tc < 1024; ++tc) mx = fmaxf(mx, bf2f(sco[row * 1024 + tc]));
        float sum = 0.0f;
        for (int tc = 0; tc < 1024; ++tc) {
            float e = __expf(bf2f(sco[row * 1024 + tc]) - mx);
            sum += e;
            sco[row * 1024 + tc] = f2bf(e);
        }
        float inv = 1.0f / sum;
        unsigned short* wf16 = (unsigned short*)wfragd;
        for (int tc = 0; tc < 1024; ++tc) {
            unsigned short wv = f2bf(bf2f(sco[row * 1024 + tc]) * inv);
            int kt = tc >> 5, kk = tc & 31;
            int la = row + ((kk >> 3) & 1) * 16;
            int va = ((kk >> 4) & 1) * 4 + ((kk & 7) >> 1);
            wf16[kt * 512 + la * 16 + va * 2 + (kk & 1)] = wv;
        }
    }
    __syncthreads();

    // ---- phase 3: attention[16 x 1024] = weights @ out
    v8f acc2[8];
#pragma unroll
    for (int j = 0; j < 8; ++j) acc2[j] = vzero();
    for (int kt = 0; kt < 32; ++kt) {
        v16bf a = ld_frag(wfragd + kt * 256, lane);
#pragma unroll
        for (int j = 0; j < 8; ++j) {
            int nt = w * 8 + j;
            v16bf bb = ld_frag(outBtm + (((size_t)b * 64 + nt) * 32 + kt) * 256, lane);
            acc2[j] = wmma_bf16(a, bb, acc2[j]);
        }
    }
#pragma unroll
    for (int j = 0; j < 8; ++j) {
        int d = (w * 8 + j) * 16 + l15;
#pragma unroll
        for (int r = 0; r < 8; ++r) {
            int s = strip * 16 + r + lh * 8;
            out[((size_t)b * SEQ + s) * 1024 + d] = acc2[j][r];
        }
    }
}

extern "C" void kernel_launch(void* const* d_in, const int* in_sizes, int n_in,
                              void* d_out, int out_size, void* d_ws, size_t ws_size,
                              hipStream_t stream) {
    const int*   ids  = (const int*)  d_in[0];
    const float* emb  = (const float*)d_in[2];
    const float* wihf = (const float*)d_in[3];
    const float* whhf = (const float*)d_in[4];
    const float* bihf = (const float*)d_in[5];
    const float* bhhf = (const float*)d_in[6];
    const float* wihb = (const float*)d_in[7];
    const float* whhb = (const float*)d_in[8];
    const float* bihb = (const float*)d_in[9];
    const float* bhhb = (const float*)d_in[10];
    float* out = (float*)d_out;
    char* ws = (char*)d_ws;

    size_t off = 0;
    unsigned* xfrag = (unsigned*)(ws + off);          off += (size_t)MT * KTE * 1024;      // 5.25 MB
    unsigned* wihfrag = (unsigned*)(ws + off);        off += (size_t)2 * NT * KTE * 1024;  // 2.62 MB
    unsigned* whhfrag = (unsigned*)(ws + off);        off += (size_t)2 * NT * KTH * 1024;  // 4.19 MB
    unsigned short* prebf = (unsigned short*)(ws + off); off += (size_t)2 * 8192 * GATES * 2; // 67 MB
    unsigned short* outA = (unsigned short*)(ws + off);  off += (size_t)8 * 64 * 32 * 1024;   // 16.8 MB
    unsigned short* outBdm = (unsigned short*)(ws + off); off += (size_t)8 * 64 * 32 * 1024;
    unsigned short* outBtm = (unsigned short*)(ws + off); off += (size_t)8 * 64 * 32 * 1024;

    pack_x_kernel<<<MT * KTE, 256, 0, stream>>>(ids, emb, xfrag);
    pack_w_kernel<<<NT * KTE, 256, 0, stream>>>(wihf, EMB, KTE, wihfrag);
    pack_w_kernel<<<NT * KTE, 256, 0, stream>>>(wihb, EMB, KTE, wihfrag + (size_t)NT * KTE * 256);
    pack_w_kernel<<<NT * KTH, 256, 0, stream>>>(whhf, HID, KTH, whhfrag);
    pack_w_kernel<<<NT * KTH, 256, 0, stream>>>(whhb, HID, KTH, whhfrag + (size_t)NT * KTH * 256);

    gemm_pre_kernel<<<dim3(MT * NT / 8, 2), 256, 0, stream>>>(xfrag, wihfrag,
                                                              bihf, bhhf, bihb, bhhb, prebf);

    lstm_scan_kernel<<<2, 1024, 0, stream>>>(prebf, whhfrag, outA, outBdm, outBtm);

    attn_kernel<<<dim3(64, 8), 256, 0, stream>>>((const unsigned*)outA, (const unsigned*)outBdm,
                                                 (const unsigned*)outBtm, out);
}
